// BNNClassifier_70849780515162
// MI455X (gfx1250) — 8-bit matrix operand order repaired and checked by reading, compiled, not run on hardware
//
#include <hip/hip_runtime.h>
#include <cstdint>
#include <cstddef>

typedef __attribute__((ext_vector_type(8))) int v8i;

#define BN_EPS 1e-5f

__global__ void binarize_input_nchw_to_nhwc(const float* __restrict__ x,
                                            int8_t* __restrict__ out,
                                            int B, int C, int H, int W) {
    size_t total = (size_t)B * C * H * W;
    size_t idx = (size_t)blockIdx.x * blockDim.x + threadIdx.x;
    if (idx >= total) return;
    int c = (int)(idx % C);
    size_t t = idx / C;
    int w = (int)(t % W); t /= W;
    int h = (int)(t % H); t /= H;
    int n = (int)t;
    float v = x[(((size_t)n * C + c) * H + h) * W + w];
    out[idx] = (v < 0.f) ? (int8_t)-1 : (int8_t)1;
}

__global__ void prep_weights(const float* __restrict__ w,
                             int8_t* __restrict__ wb,
                             int OC, int C, int KPAD) {
    int K = 9 * C;
    int idx = blockIdx.x * blockDim.x + threadIdx.x;
    if (idx >= OC * KPAD) return;
    int k  = idx % KPAD;
    int oc = idx / KPAD;
    int8_t v = 0;
    if (k < K) {
        int tap = k / C;
        int c   = k - tap * C;
        float f = w[((size_t)oc * C + c) * 9 + tap];
        v = (f < 0.f) ? (int8_t)-1 : (int8_t)1;
    }
    wb[idx] = v;
}

__global__ void zero_f32(float* __restrict__ p, int n) {
    int i = blockIdx.x * blockDim.x + threadIdx.x;
    if (i < n) p[i] = 0.f;
}

template <int KPAD>
__global__ __launch_bounds__(256)
void conv_bin_wmma(const int8_t* __restrict__ in,
                   const int8_t* __restrict__ wgt,
                   int16_t* __restrict__ out,
                   float* __restrict__ chSum,
                   float* __restrict__ chSumSq,
                   int B, int H, int W, int C,
                   int OH, int OW, int OC, int stride) {
    __shared__ int8_t ldsA[64 * KPAD];
    __shared__ float  ldsSum[32];
    __shared__ float  ldsSq[32];

    const int tid = threadIdx.x;
    const int K = 9 * C;
    const size_t mBase = (size_t)blockIdx.x * 64;

    if (tid < 32) { ldsSum[tid] = 0.f; ldsSq[tid] = 0.f; }

    if ((C & 3) == 0) {
        for (int e = tid * 4; e < 64 * KPAD; e += 256 * 4) {
            int row = e / KPAD;
            int kb  = e - row * KPAD;
            uint32_t val = 0u;
            if (kb < K) {
                int tap = kb / C;
                int c   = kb - tap * C;
                size_t m = mBase + row;
                int ow = (int)(m % OW);
                size_t t = m / OW;
                int oh = (int)(t % OH);
                int n  = (int)(t / OH);
                int ih = oh * stride - 1 + tap / 3;
                int iw = ow * stride - 1 + (tap - (tap / 3) * 3);
                if ((unsigned)ih < (unsigned)H && (unsigned)iw < (unsigned)W)
                    val = *(const uint32_t*)&in[(((size_t)n * H + ih) * W + iw) * C + c];
            }
            *(uint32_t*)&ldsA[e] = val;
        }
    } else {
        for (int e = tid * 4; e < 64 * KPAD; e += 256 * 4) {
            int row = e / KPAD;
            int kb  = e - row * KPAD;
            size_t m = mBase + row;
            int ow = (int)(m % OW);
            size_t t = m / OW;
            int oh = (int)(t % OH);
            int n  = (int)(t / OH);
            uint32_t val = 0u;
            for (int b = 0; b < 4; ++b) {
                int k = kb + b;
                if (k < K) {
                    int tap = k / C;
                    int c   = k - tap * C;
                    int ih = oh * stride - 1 + tap / 3;
                    int iw = ow * stride - 1 + (tap - (tap / 3) * 3);
                    if ((unsigned)ih < (unsigned)H && (unsigned)iw < (unsigned)W) {
                        uint8_t bv = (uint8_t)in[(((size_t)n * H + ih) * W + iw) * C + c];
                        val |= (uint32_t)bv << (8 * b);
                    }
                }
            }
            *(uint32_t*)&ldsA[e] = val;
        }
    }
    __syncthreads();

    const int wid  = tid >> 5;
    const int lane = tid & 31;
    const int wm   = wid & 3;
    const int wn   = wid >> 2;
    const int half = lane >> 4;
    const int lm   = lane & 15;

    const int ocBase = blockIdx.y * 32 + wn * 16;
    const int8_t* wRow = wgt + (size_t)(ocBase + lm) * KPAD;
    const int8_t* aRow = &ldsA[(wm * 16 + lm) * KPAD];

    __builtin_prefetch(wRow, 0, 1);

    v8i acc = {};
#pragma unroll
    for (int k0 = 0; k0 < KPAD; k0 += 64) {
        int2 a01 = *(const int2*)(aRow + k0 +      half * 16);
        int2 a23 = *(const int2*)(aRow + k0 +  8 + half * 16);
        int2 a45 = *(const int2*)(aRow + k0 + 32 + half * 16);
        int2 a67 = *(const int2*)(aRow + k0 + 40 + half * 16);
        v8i a;
        a[0] = a01.x; a[1] = a01.y; a[2] = a23.x; a[3] = a23.y;
        a[4] = a45.x; a[5] = a45.y; a[6] = a67.x; a[7] = a67.y;
        int4 b03 = *(const int4*)(wRow + k0 +      half * 16);
        int4 b47 = *(const int4*)(wRow + k0 + 32 + half * 16);
        v8i b;
        b[0] = b03.x; b[1] = b03.y; b[2] = b03.z; b[3] = b03.w;
        b[4] = b47.x; b[5] = b47.y; b[6] = b47.z; b[7] = b47.w;
        acc = __builtin_amdgcn_wmma_i32_16x16x64_iu8(true, a, true, b, acc,
                                                     false, false);
    }

    float s = 0.f, q = 0.f;
    const size_t mg0 = mBase + (size_t)wm * 16 + half * 8;
    const int ocg = ocBase + lm;
#pragma unroll
    for (int v = 0; v < 8; ++v) {
        int d = acc[v];
        out[(mg0 + v) * OC + ocg] = (int16_t)d;
        float fd = (float)d;
        s += fd; q += fd * fd;
    }
    atomicAdd(&ldsSum[wn * 16 + lm], s);
    atomicAdd(&ldsSq [wn * 16 + lm], q);
    __syncthreads();
    if (tid < 32) {
        atomicAdd(&chSum  [blockIdx.y * 32 + tid], ldsSum[tid]);
        atomicAdd(&chSumSq[blockIdx.y * 32 + tid], ldsSq[tid]);
    }
}

__global__ void bn_binarize(const int16_t* __restrict__ y,
                            const float* __restrict__ chSum,
                            const float* __restrict__ chSumSq,
                            const float* __restrict__ gamma,
                            const float* __restrict__ beta,
                            int8_t* __restrict__ out,
                            size_t M, int OC) {
    size_t idx = (size_t)blockIdx.x * blockDim.x + threadIdx.x;
    if (idx >= M * (size_t)OC) return;
    int c = (int)(idx % OC);
    float invM = 1.f / (float)M;
    float mean = chSum[c] * invM;
    float var  = chSumSq[c] * invM - mean * mean;
    float t = gamma[c] * rsqrtf(var + BN_EPS);
    float z = t * ((float)y[idx] - mean) + beta[c];
    out[idx] = (z < 0.f) ? (int8_t)-1 : (int8_t)1;
}

__global__ void avgpool(const int8_t* __restrict__ h,
                        float* __restrict__ pooled,
                        int B, int HW, int C) {
    int idx = blockIdx.x * blockDim.x + threadIdx.x;
    if (idx >= B * C) return;
    int c = idx % C;
    int n = idx / C;
    const int8_t* p = h + (size_t)n * HW * C + c;
    int s = 0;
    for (int i = 0; i < HW; ++i) s += (int)p[(size_t)i * C];
    pooled[idx] = (float)s / (float)HW;
}

__global__ void fc_head(const float* __restrict__ pooled,
                        const float* __restrict__ w,
                        const float* __restrict__ bias,
                        float* __restrict__ out,
                        int B, int CIN, int NC) {
    int n = blockIdx.x * blockDim.x + threadIdx.x;
    if (n >= B) return;
    for (int j = 0; j < NC; ++j) {
        float acc = bias[j];
        for (int k = 0; k < CIN; ++k)
            acc += pooled[n * CIN + k] * w[j * CIN + k];
        out[n * NC + j] = acc;
    }
}

extern "C" void kernel_launch(void* const* d_in, const int* in_sizes, int n_in,
                              void* d_out, int out_size, void* d_ws, size_t ws_size,
                              hipStream_t stream) {
    (void)in_sizes; (void)n_in; (void)out_size; (void)ws_size;

    const float* x    = (const float*)d_in[0];
    const float* w1   = (const float*)d_in[1];
    const float* g1   = (const float*)d_in[2];
    const float* b1   = (const float*)d_in[3];
    const float* w2   = (const float*)d_in[4];
    const float* g2   = (const float*)d_in[5];
    const float* b2   = (const float*)d_in[6];
    const float* w3   = (const float*)d_in[7];
    const float* g3   = (const float*)d_in[8];
    const float* b3   = (const float*)d_in[9];
    const float* fcw  = (const float*)d_in[10];
    const float* fcb  = (const float*)d_in[11];

    const int B = 64;
    const int H1 = 224, W1 = 224, C1 = 3,  OC1 = 32,  S1 = 1, OH1 = 224, OW1 = 224;
    const int H2 = 224, W2 = 224, C2 = 32, OC2 = 64,  S2 = 2, OH2 = 112, OW2 = 112;
    const int H3 = 112, W3 = 112, C3 = 64, OC3 = 128, S3 = 2, OH3 = 56,  OW3 = 56;
    const size_t M1 = (size_t)B * OH1 * OW1;
    const size_t M2 = (size_t)B * OH2 * OW2;
    const size_t M3 = (size_t)B * OH3 * OW3;

    char* ws = (char*)d_ws;
    const size_t capBin  = 102760448;
    const size_t capConv = 205520896;
    int8_t*  binA   = (int8_t*)(ws);
    int16_t* convB  = (int16_t*)(ws + capBin);
    int8_t*  binB   = (int8_t*)(ws + capBin + capConv);
    float*   chSum  = (float*)(ws + capBin + capConv + capBin);
    float*   chSq   = chSum + 128;
    float*   pooled = chSum + 256;
    int8_t*  w1b    = (int8_t*)(pooled + 64 * 128);
    int8_t*  w2b    = w1b + 32 * 64;
    int8_t*  w3b    = w2b + 64 * 320;

    {
        size_t total = (size_t)B * C1 * H1 * W1;
        binarize_input_nchw_to_nhwc<<<(unsigned)((total + 255) / 256), 256, 0, stream>>>(
            x, binA, B, C1, H1, W1);
    }
    prep_weights<<<(32 * 64 + 255) / 256, 256, 0, stream>>>(w1, w1b, OC1, C1, 64);
    prep_weights<<<(64 * 320 + 255) / 256, 256, 0, stream>>>(w2, w2b, OC2, C2, 320);
    prep_weights<<<(128 * 576 + 255) / 256, 256, 0, stream>>>(w3, w3b, OC3, C3, 576);

    zero_f32<<<1, 256, 0, stream>>>(chSum, 256);
    conv_bin_wmma<64><<<dim3((unsigned)(M1 / 64), OC1 / 32), 256, 0, stream>>>(
        binA, w1b, convB, chSum, chSq, B, H1, W1, C1, OH1, OW1, OC1, S1);
    bn_binarize<<<(unsigned)((M1 * OC1 + 255) / 256), 256, 0, stream>>>(
        convB, chSum, chSq, g1, b1, binB, M1, OC1);

    zero_f32<<<1, 256, 0, stream>>>(chSum, 256);
    conv_bin_wmma<320><<<dim3((unsigned)(M2 / 64), OC2 / 32), 256, 0, stream>>>(
        binB, w2b, convB, chSum, chSq, B, H2, W2, C2, OH2, OW2, OC2, S2);
    bn_binarize<<<(unsigned)((M2 * OC2 + 255) / 256), 256, 0, stream>>>(
        convB, chSum, chSq, g2, b2, binA, M2, OC2);

    zero_f32<<<1, 256, 0, stream>>>(chSum, 256);
    conv_bin_wmma<576><<<dim3((unsigned)(M3 / 64), OC3 / 32), 256, 0, stream>>>(
        binA, w3b, convB, chSum, chSq, B, H3, W3, C3, OH3, OW3, OC3, S3);
    bn_binarize<<<(unsigned)((M3 * OC3 + 255) / 256), 256, 0, stream>>>(
        convB, chSum, chSq, g3, b3, binB, M3, OC3);

    avgpool<<<(B * OC3 + 255) / 256, 256, 0, stream>>>(binB, pooled, B, OH3 * OW3, OC3);
    fc_head<<<1, 64, 0, stream>>>(pooled, fcw, fcb, (float*)d_out, B, OC3, 2);
}
